// TGVAE_78855599555056
// MI455X (gfx1250) — matrix operand order repaired and checked by reading, not compiled, not run on hardware
//
#include <hip/hip_runtime.h>
#include <hip/hip_bf16.h>
#include <math.h>

#define DEV __device__ __forceinline__

namespace {

constexpr int Nn_   = 16384;   // nodes
constexpr int Ee_   = 65536;   // edges
constexpr int Bb_   = 512;     // batch
constexpr int Ll_   = 64;      // seq len
constexpr int Dd_   = 512;
constexpr int DFF_  = 1024;
constexpr int EHh_  = 16;      // GAT heads
constexpr int DKk_  = 32;      // decoder head dim
constexpr int DHh_  = 16;      // decoder heads
constexpr int NL_   = 8;
constexpr int SMIV_ = 100;

typedef __attribute__((ext_vector_type(16))) __bf16 v16bf;
typedef __attribute__((ext_vector_type(8)))  float  v8f;

DEV unsigned short f2bf(float f) {           // native RNE float -> bf16
  __bf16 h = (__bf16)f;
  return __builtin_bit_cast(unsigned short, h);
}
DEV float leakyf(float x, float sl) { return x > 0.f ? x : sl * x; }

DEV void atomicMaxF(float* addr, float val) {
  int* ia = (int*)addr;
  int cur = __float_as_int(*addr);
  while (__int_as_float(cur) < val) {
    int old = atomicCAS(ia, cur, __float_as_int(val));
    if (old == cur) break;
    cur = old;
  }
}
DEV float waveSum(float v) {
  #pragma unroll
  for (int o = 16; o; o >>= 1) v += __shfl_xor(v, o, 32);
  return v;
}
DEV uint4 pack8(const unsigned short t[8]) {
  uint4 r;
  r.x = (unsigned)t[0] | ((unsigned)t[1] << 16);
  r.y = (unsigned)t[2] | ((unsigned)t[3] << 16);
  r.z = (unsigned)t[4] | ((unsigned)t[5] << 16);
  r.w = (unsigned)t[6] | ((unsigned)t[7] << 16);
  return r;
}

// ---------------------------------------------------------------------------
// Weight pre-pass: fp32 W[K,N] -> bf16, swizzled into WMMA B-fragment tile
// order and zero-padded in N. Tile = 64 cols x 32 K = 2048 elems laid out as
// [sub(4)][lane(32)][elem(16)], blocks ordered [ntile][ktile] so a GEMM block
// streams K*64 contiguous bf16.  Frag map: lane=(Nloc&15)+16*(Kloc>>4),
// elem=Kloc&15, sub=Nloc>>4.
// ---------------------------------------------------------------------------
__global__ void k_wswz(const float* __restrict__ Wm, unsigned short* __restrict__ out,
                       int K, int Nc, int NcPad)
{
  long i = (long)blockIdx.x * blockDim.x + threadIdx.x;   // over NcPad*K
  if (i >= (long)NcPad * K) return;
  int idx  = (int)(i & 2047);
  long blk = i >> 11;
  int kt = (int)(blk % (K >> 5));
  int nt = (int)(blk / (K >> 5));
  int sub  = idx >> 9;
  int lane = (idx >> 4) & 31;
  int elem = idx & 15;
  int kk = kt * 32 + (lane >> 4) * 16 + elem;
  int nn = nt * 64 + sub * 16 + (lane & 15);
  out[i] = (nn < Nc) ? f2bf(Wm[(long)kk * Nc + nn]) : (unsigned short)0;
}

// ---------------------------------------------------------------------------
// WMMA bf16 GEMM:  C[M,N] = act( A[M,K] @ W[K,N] + bias )   (optional +=)
// A fp32 [M,K]; W pre-swizzled bf16 (see k_wswz). block=256 (8 waves),
// block tile 64x64, k-step 32. wave: wm=wave>>1 row strip, wn=wave&1 col half.
// A tile staged via fp32->bf16 convert into fragment-order LDS (1 b128 store
// per thread); B tile staged with GLOBAL_LOAD_ASYNC_TO_LDS_B128 (pure 16-bit
// copy, ASYNCcnt-tracked). Fragments then read as aligned ds_load_b128 pairs.
// All M multiples of 64, K of 32; NGUARD guards only the N=100 epilogue.
// ---------------------------------------------------------------------------
template <int ACT, bool ACCUM, bool NGUARD>
__global__ __launch_bounds__(256)
void k_gemm(const float* __restrict__ A, const unsigned short* __restrict__ Wswz,
            const float* __restrict__ bias, float* __restrict__ Cmat,
            int M, int Nc, int K)
{
  __shared__ alignas(16) unsigned short Afrag[4][32][16];  // 4KB
  __shared__ alignas(16) unsigned short Bfrag[4][32][16];  // 4KB

  const int tid  = threadIdx.x;
  const int lane = tid & 31;
  const int wave = tid >> 5;
  const int wm   = wave >> 1;   // 0..3 : 16-row strip
  const int wn   = wave & 1;    // 0..1 : 32-col half

  const int mBlock = blockIdx.y * 64;
  const int nBlock = blockIdx.x * 64;

  v8f acc0 = {0.f,0.f,0.f,0.f,0.f,0.f,0.f,0.f};
  v8f acc1 = acc0;

  // A staging: thread -> row arow (0..63), 8 consecutive K at acol
  //   frag map: kh=(K>>4)&1, elem=K&15, lane=(M&15)+16*kh
  const int arow   = tid >> 2;
  const int acol   = (tid & 3) * 8;            // 0,8,16,24
  const int aStrip = arow >> 4;
  const int aLane  = (arow & 15) + ((acol >> 4) & 1) * 16;
  const int aElem  = (acol & 8) ? 8 : 0;
  const float* aptr = A + (long)(mBlock + arow) * K + acol;

  // B staging: thread copies 8 contiguous bf16 of the swizzled tile
  const unsigned short* wptr = Wswz + (long)blockIdx.x * ((long)K * 64) + (long)tid * 8;
  const unsigned ldsB = (unsigned)(size_t)(&Bfrag[0][0][0]) + (unsigned)tid * 16;

  for (int k0 = 0; k0 < K; k0 += 32) {
    { // stage A: 2x b128 global load, native cvt, 1x b128 DS store
      const float4* ap = (const float4*)(aptr + k0);
      float4 f0 = ap[0], f1 = ap[1];
      unsigned short t[8] = { f2bf(f0.x), f2bf(f0.y), f2bf(f0.z), f2bf(f0.w),
                              f2bf(f1.x), f2bf(f1.y), f2bf(f1.z), f2bf(f1.w) };
      *(uint4*)&Afrag[aStrip][aLane][aElem] = pack8(t);
    }
    { // stage B: async global->LDS b128 copy (no VGPR round-trip)
      unsigned long long ga = (unsigned long long)(size_t)(wptr + ((long)(k0 >> 5) << 11));
      asm volatile("global_load_async_to_lds_b128 %0, %1, off"
                   :: "v"(ldsB), "v"(ga) : "memory");
    }
    if (k0 + 32 < K) {  // gfx1250 global_prefetch of next K tile
      __builtin_prefetch(aptr + k0 + 32, 0, 1);
      __builtin_prefetch(wptr + ((long)((k0 + 32) >> 5) << 11), 0, 1);
    }
    asm volatile("s_wait_asynccnt 0" ::: "memory");
    __syncthreads();

    union { uint4 q[2]; v16bf v; } af, b0, b1;
    af.q[0] = *(const uint4*)&Afrag[wm][lane][0];
    af.q[1] = *(const uint4*)&Afrag[wm][lane][8];
    b0.q[0] = *(const uint4*)&Bfrag[wn * 2][lane][0];
    b0.q[1] = *(const uint4*)&Bfrag[wn * 2][lane][8];
    b1.q[0] = *(const uint4*)&Bfrag[wn * 2 + 1][lane][0];
    b1.q[1] = *(const uint4*)&Bfrag[wn * 2 + 1][lane][8];

    acc0 = __builtin_amdgcn_wmma_f32_16x16x32_bf16(false, af.v, false, b0.v,
                                                   (short)0, acc0, false, false);
    acc1 = __builtin_amdgcn_wmma_f32_16x16x32_bf16(false, af.v, false, b1.v,
                                                   (short)0, acc1, false, false);
    __syncthreads();
  }

  // epilogue: C VGPR r -> row = Moff + (lane>>4)*8 + r, col = Noff + (lane&15)
  const int orow0 = mBlock + wm * 16 + (lane >> 4) * 8;
  const int ocol  = nBlock + wn * 32 + (lane & 15);
  #pragma unroll
  for (int r = 0; r < 8; ++r) {
    const int row = orow0 + r;
    #pragma unroll
    for (int t = 0; t < 2; ++t) {
      const int col = ocol + t * 16;
      if (NGUARD && col >= Nc) continue;
      float v = t ? acc1[r] : acc0[r];
      if (bias) v += bias[col];
      if (ACT == 1) v = v > 0.f ? v : 0.f;
      if (ACT == 2) v = leakyf(v, 0.01f);
      long idx = (long)row * Nc + col;
      Cmat[idx] = ACCUM ? (Cmat[idx] + v) : v;
    }
  }
}

// ---------------------------------------------------------------------------
__global__ void k_fill(float* p, float v, long n) {
  long i = (long)blockIdx.x * blockDim.x + threadIdx.x;
  if (i < n) p[i] = v;
}

// initial GAT: xp = ne.W[node_feature], s/d dot products (H=1, C=D)
__global__ __launch_bounds__(256)
void k_ne_gather(const int* __restrict__ nodef, const float* __restrict__ neW,
                 const float* __restrict__ asrc, const float* __restrict__ adst,
                 float* __restrict__ xp, float* __restrict__ s, float* __restrict__ d)
{
  __shared__ float r1[256], r2[256];
  const int n = blockIdx.x, t = threadIdx.x;
  const int f = nodef[n];
  float s1 = 0.f, s2 = 0.f;
  for (int k = t; k < Dd_; k += 256) {
    float v = neW[(long)f * Dd_ + k];
    xp[(long)n * Dd_ + k] = v;
    s1 += v * asrc[k];
    s2 += v * adst[k];
  }
  r1[t] = s1; r2[t] = s2; __syncthreads();
  for (int o = 128; o; o >>= 1) {
    if (t < o) { r1[t] += r1[t + o]; r2[t] += r2[t + o]; }
    __syncthreads();
  }
  if (t == 0) { s[n] = r1[0]; d[n] = r2[0]; }
}

// per-head src/dst scores for H=16, C=32 (one wave per head)
__global__ __launch_bounds__(512)
void k_heads_sd(const float* __restrict__ xp, const float* __restrict__ as,
                const float* __restrict__ ad, float* __restrict__ s, float* __restrict__ d)
{
  const int n = blockIdx.x, h = threadIdx.x >> 5, c = threadIdx.x & 31;
  float xv = xp[(long)n * Dd_ + h * 32 + c];
  float sv = waveSum(xv * as[h * 32 + c]);
  float dv = waveSum(xv * ad[h * 32 + c]);
  if (c == 0) { s[(long)n * EHh_ + h] = sv; d[(long)n * EHh_ + h] = dv; }
}

// per-edge-type attention score table: esc[v,h] = sum_c We[v,h*32+c]*a_edge[h,c]
__global__ void k_escore(const float* __restrict__ We, const float* __restrict__ ae,
                         float* __restrict__ esc) {
  int i = threadIdx.x;
  if (i >= 64) return;
  int v = i >> 4, h = i & 15;
  float s = 0.f;
  for (int c = 0; c < 32; ++c) s += We[v * Dd_ + h * 32 + c] * ae[h * 32 + c];
  esc[v * EHh_ + h] = s;
}

__global__ void k_deg(const int* __restrict__ dst, const int* __restrict__ ea,
                      float* __restrict__ deg, float* __restrict__ cnt, int E) {
  int e = blockIdx.x * blockDim.x + threadIdx.x;
  if (e >= E) return;
  atomicAdd(&deg[dst[e]], 1.f);
  atomicAdd(&cnt[dst[e] * 4 + ea[e]], 1.f);
}
__global__ void k_loopattr(float* la, const float* deg, long n4) {
  long i = (long)blockIdx.x * blockDim.x + threadIdx.x;
  if (i < n4) la[i] = la[i] / fmaxf(deg[i >> 2], 1.f);
}

__global__ void k_alpha(const float* __restrict__ s, const float* __restrict__ d,
                        const int* __restrict__ src, const int* __restrict__ dst,
                        const int* __restrict__ ea, const float* __restrict__ esc,
                        const float* __restrict__ la, float* __restrict__ alpha,
                        int H, int nE, long tot)
{
  long i = (long)blockIdx.x * blockDim.x + threadIdx.x;
  if (i >= tot) return;
  int h = (int)(i % H);
  long e = i / H;
  int sf, df; float ext = 0.f;
  if (e < nE) {
    sf = src[e]; df = dst[e];
    if (esc) ext = esc[ea[e] * EHh_ + h];
  } else {
    int n = (int)(e - nE); sf = df = n;
    if (esc) {
      #pragma unroll
      for (int v = 0; v < 4; ++v) ext += la[(long)n * 4 + v] * esc[v * EHh_ + h];
    }
  }
  float a = s[(long)sf * H + h] + d[(long)df * H + h] + ext;
  alpha[i] = leakyf(a, 0.2f);
}

__global__ void k_segmax(const float* __restrict__ alpha, const int* __restrict__ dst,
                         float* __restrict__ mx, int H, int nE, long tot) {
  long i = (long)blockIdx.x * blockDim.x + threadIdx.x;
  if (i >= tot) return;
  int h = (int)(i % H);
  long e = i / H;
  int df = (e < nE) ? dst[e] : (int)(e - nE);
  atomicMaxF(&mx[(long)df * H + h], alpha[i]);
}

__global__ void k_expsum(float* __restrict__ alpha, const int* __restrict__ dst,
                         const float* __restrict__ mx, float* __restrict__ den,
                         int H, int nE, long tot) {
  long i = (long)blockIdx.x * blockDim.x + threadIdx.x;
  if (i >= tot) return;
  int h = (int)(i % H);
  long e = i / H;
  int df = (e < nE) ? dst[e] : (int)(e - nE);
  float w = __expf(alpha[i] - mx[(long)df * H + h]);
  alpha[i] = w;
  atomicAdd(&den[(long)df * H + h], w);
}

// out[df, :] += (w/den) * xp[sf, :]
__global__ __launch_bounds__(256)
void k_scatter(const float* __restrict__ alpha, const float* __restrict__ den,
               const int* __restrict__ src, const int* __restrict__ dst,
               const float* __restrict__ xp, float* __restrict__ out, int H, int nE)
{
  const int e = blockIdx.x;
  int sf, df;
  if (e < nE) { sf = src[e]; df = dst[e]; }
  else        { sf = df = e - nE; }
  #pragma unroll
  for (int r = 0; r < 2; ++r) {
    int k = threadIdx.x + r * 256;
    int h = (k * H) >> 9;                 // k / (D/H)
    float w = alpha[(long)e * H + h];
    float dn = den[(long)df * H + h] + 1e-16f;
    atomicAdd(&out[(long)df * Dd_ + k], (w / dn) * xp[(long)sf * Dd_ + k]);
  }
}

__global__ void k_bias_act(const float* __restrict__ g, const float* __restrict__ bias,
                           float* __restrict__ out, long n) {
  long i = (long)blockIdx.x * blockDim.x + threadIdx.x;
  if (i >= n) return;
  out[i] = leakyf(g[i] + bias[i % Dd_], 0.01f);
}

// LayerNorm (sample std, /(D-1), eps on std), optional pre-bias / act / residual
// MODE 0: out = LN(x+preb) ; MODE 1: out = res + act(LN(x+preb))
template <int ACT, int MODE>
__global__ __launch_bounds__(256)
void k_ln(const float* __restrict__ X, const float* __restrict__ preb,
          const float* __restrict__ g, const float* __restrict__ bvec,
          const float* __restrict__ res, float* __restrict__ out)
{
  __shared__ float r1[256], r2[256];
  const long row = blockIdx.x;
  const int t = threadIdx.x;
  const float* x = X + row * Dd_;
  float a0 = x[t]       + (preb ? preb[t]       : 0.f);
  float a1 = x[t + 256] + (preb ? preb[t + 256] : 0.f);
  r1[t] = a0 + a1; r2[t] = a0 * a0 + a1 * a1;
  __syncthreads();
  for (int o = 128; o; o >>= 1) {
    if (t < o) { r1[t] += r1[t + o]; r2[t] += r2[t + o]; }
    __syncthreads();
  }
  float sum = r1[0], sq = r2[0];
  float m   = sum / Dd_;
  float var = (sq - sum * sum / Dd_) / (Dd_ - 1);
  float inv = 1.f / (sqrtf(fmaxf(var, 0.f)) + 1e-6f);
  float y0 = g[t]       * (a0 - m) * inv + bvec[t];
  float y1 = g[t + 256] * (a1 - m) * inv + bvec[t + 256];
  if (ACT == 2) { y0 = leakyf(y0, 0.01f); y1 = leakyf(y1, 0.01f); }
  if (MODE == 1) { y0 += res[row * Dd_ + t]; y1 += res[row * Dd_ + t + 256]; }
  out[row * Dd_ + t] = y0;
  out[row * Dd_ + t + 256] = y1;
}

// encoder scan carry: nf' = h + dense ; dense' = nf' + dense
__global__ void k_carry(float* __restrict__ nf, float* __restrict__ dense,
                        const float* __restrict__ h, long n) {
  long i = (long)blockIdx.x * blockDim.x + threadIdx.x;
  if (i >= n) return;
  float dn = dense[i];
  float nf2 = h[i] + dn;
  nf[i] = nf2;
  dense[i] = nf2 + dn;
}

// batch = repeat(arange(B), N/B) -> contiguous 32-row sums
__global__ void k_pool(const float* __restrict__ nf, float* __restrict__ pool) {
  int b = blockIdx.x;
  for (int d = threadIdx.x; d < Dd_; d += blockDim.x) {
    float s = 0.f;
    #pragma unroll 4
    for (int i = 0; i < 32; ++i) s += nf[(long)(b * 32 + i) * Dd_ + d];
    pool[(long)b * Dd_ + d] = s;
  }
}

__global__ void k_z(const float* __restrict__ mu, const float* __restrict__ sg,
                    const float* __restrict__ eps, float* __restrict__ z, long n) {
  long i = (long)blockIdx.x * blockDim.x + threadIdx.x;
  if (i < n) z[i] = mu[i] + __expf(0.5f * sg[i]) * eps[i];
}

__global__ void k_dec_init(const int* __restrict__ smi, const float* __restrict__ emb,
                           float* __restrict__ xd) {
  long i = (long)blockIdx.x * blockDim.x + threadIdx.x;
  if (i >= (long)Bb_ * Ll_ * Dd_) return;
  int d = (int)(i % Dd_);
  long bl = i / Dd_;
  int l = (int)(bl % Ll_);
  int tok = smi[bl];
  int ip = d >> 1;
  float div = __expf((float)(2 * ip) * (-9.210340371976184f / (float)Dd_));
  float pe = (d & 1) ? __cosf((float)l * div) : __sinf((float)l * div);
  xd[i] = emb[(long)tok * Dd_ + d] * 22.62741699796952f + pe;
}

// causal self-attention: one wave per (b,h,q); lane = head-dim element
__global__ __launch_bounds__(256)
void k_attn(const float* __restrict__ Q, const float* __restrict__ Kk,
            const float* __restrict__ V, float* __restrict__ O)
{
  __shared__ float sc[8][Ll_];
  const int wave = threadIdx.x >> 5, lane = threadIdx.x & 31;
  long gw = (long)blockIdx.x * 8 + wave;
  if (gw >= (long)Bb_ * DHh_ * Ll_) return;
  const int q = (int)(gw % Ll_);
  long t = gw / Ll_;
  const int h = (int)(t % DHh_);
  const int b = (int)(t / DHh_);
  const float scale = 0.17677669529663687f;  // 1/sqrt(32)
  const float qd = Q[((long)b * Ll_ + q) * Dd_ + h * DKk_ + lane];
  float mx = -1e30f;
  for (int k = 0; k <= q; ++k) {
    float p = waveSum(qd * Kk[((long)b * Ll_ + k) * Dd_ + h * DKk_ + lane]) * scale;
    sc[wave][k] = p;              // same value in all lanes
    mx = fmaxf(mx, p);
  }
  float w0 = (lane      <= q) ? __expf(sc[wave][lane]      - mx) : 0.f;
  float w1 = (lane + 32 <= q) ? __expf(sc[wave][lane + 32] - mx) : 0.f;
  float den = waveSum(w0 + w1);
  float o = 0.f;
  for (int k = 0; k <= q; ++k)
    o += (__expf(sc[wave][k] - mx) / den) *
         V[((long)b * Ll_ + k) * Dd_ + h * DKk_ + lane];
  O[((long)b * Ll_ + q) * Dd_ + h * DKk_ + lane] = o;
}

// cross-attention output broadcast: xd[b,l,:] += co[b,:]
__global__ void k_bcastadd(float* __restrict__ xd, const float* __restrict__ co, long n) {
  long i = (long)blockIdx.x * blockDim.x + threadIdx.x;
  if (i >= n) return;
  int d = (int)(i % Dd_);
  long b = i / ((long)Ll_ * Dd_);
  xd[i] += co[b * Dd_ + d];
}

__global__ __launch_bounds__(128)
void k_lsm(const float* __restrict__ logits, float* __restrict__ out) {
  __shared__ float red[128];
  const long row = blockIdx.x;
  const int t = threadIdx.x;
  float v = (t < SMIV_) ? logits[row * SMIV_ + t] : -1e30f;
  red[t] = v; __syncthreads();
  for (int o = 64; o; o >>= 1) { if (t < o) red[t] = fmaxf(red[t], red[t + o]); __syncthreads(); }
  float mx = red[0]; __syncthreads();
  float e = (t < SMIV_) ? __expf(v - mx) : 0.f;
  red[t] = e; __syncthreads();
  for (int o = 64; o; o >>= 1) { if (t < o) red[t] += red[t + o]; __syncthreads(); }
  float ls = logf(red[0]);
  if (t < SMIV_) out[row * SMIV_ + t] = v - mx - ls;
}

inline int cdiv(long a, long b) { return (int)((a + b - 1) / b); }

} // namespace

// ===========================================================================
extern "C" void kernel_launch(void* const* d_in, const int* in_sizes, int n_in,
                              void* d_out, int out_size, void* d_ws, size_t ws_size,
                              hipStream_t stream)
{
  (void)in_sizes; (void)n_in; (void)out_size; (void)ws_size;

  // ---- inputs (setup_inputs order, then params pytree leaves, dicts sorted) --
  const int*   nodef = (const int*)d_in[0];
  const int*   srcI  = (const int*)d_in[1];
  const int*   dstI  = srcI + Ee_;
  const int*   eattr = (const int*)d_in[2];
  const int*   smi   = (const int*)d_in[4];
  const float* eps   = (const float*)d_in[6];

  const float* dec_caW = (const float*)d_in[7];   // [NL,4,D,D]
  const float* dec_cab = (const float*)d_in[8];   // [NL,4,D]
  const float* dec_fb1 = (const float*)d_in[9];
  const float* dec_fb2 = (const float*)d_in[10];
  const float* dec_fw1 = (const float*)d_in[11];  // [NL,D,DFF]
  const float* dec_fw2 = (const float*)d_in[12];  // [NL,DFF,D]
  const float* dec_lnb = (const float*)d_in[13];  // [NL,3,D]
  const float* dec_lng = (const float*)d_in[14];
  const float* dec_saW = (const float*)d_in[15];
  const float* dec_sab = (const float*)d_in[16];
  const float* decLNb  = (const float*)d_in[17];
  const float* decLNg  = (const float*)d_in[18];
  const float* embW    = (const float*)d_in[19];  // [SMI_V,D]
  const float* enc_fb1 = (const float*)d_in[20];
  const float* enc_fb2 = (const float*)d_in[21];
  const float* enc_fw1 = (const float*)d_in[22];
  const float* enc_fw2 = (const float*)d_in[23];
  const float* g1W  = (const float*)d_in[24];     // [NL,D,D]
  const float* g1We = (const float*)d_in[25];     // [NL,4,D]
  const float* g1ad = (const float*)d_in[26];     // [NL,16,32]
  const float* g1ae = (const float*)d_in[27];
  const float* g1as = (const float*)d_in[28];
  const float* g1b  = (const float*)d_in[29];     // [NL,D]
  const float* g2W  = (const float*)d_in[30];
  const float* g2We = (const float*)d_in[31];
  const float* g2ad = (const float*)d_in[32];
  const float* g2ae = (const float*)d_in[33];
  const float* g2as = (const float*)d_in[34];
  const float* g2b  = (const float*)d_in[35];
  const float* enc_lnb = (const float*)d_in[36];
  const float* enc_lng = (const float*)d_in[37];
  const float* encLNb  = (const float*)d_in[38];
  const float* encLNg  = (const float*)d_in[39];
  const float* genW = (const float*)d_in[40];     // [D,SMI_V]
  const float* genb = (const float*)d_in[41];
  const float* muW  = (const float*)d_in[42];
  const float* mub  = (const float*)d_in[43];
  const float* neW  = (const float*)d_in[44];     // [NODE_V,D]
  const float* nead = (const float*)d_in[45];
  const float* neas = (const float*)d_in[46];
  const float* neb  = (const float*)d_in[47];
  const float* sgW  = (const float*)d_in[48];
  const float* sgb  = (const float*)d_in[49];

  float* outp  = (float*)d_out;                       // [B,L,100]
  float* muOut = outp + (long)Bb_ * Ll_ * SMIV_;      // [B,D]
  float* sgOut = muOut + (long)Bb_ * Dd_;             // [B,D]

  // ---- workspace (encoder/decoder overlay) --------------------------------
  float* W = (float*)d_ws;
  size_t o = 0;
  auto alloc = [&](size_t n) { float* p = W + o; o += n; return p; };
  float* zB    = alloc((size_t)Bb_ * Dd_);
  float* poolB = alloc((size_t)Bb_ * Dd_);
  unsigned short* wbfB = (unsigned short*)alloc((size_t)Dd_ * DFF_ / 2); // bf16 swizzled weights
  const size_t obase = o;
  // encoder view
  float* nfB    = alloc((size_t)Nn_ * Dd_);
  float* denseB = alloc((size_t)Nn_ * Dd_);
  float* hB     = alloc((size_t)Nn_ * Dd_);
  float* xpB    = alloc((size_t)Nn_ * Dd_);
  float* gatoB  = alloc((size_t)Nn_ * Dd_);
  float* t0B    = alloc((size_t)Nn_ * DFF_);
  float* sB     = alloc((size_t)Nn_ * EHh_);
  float* dB     = alloc((size_t)Nn_ * EHh_);
  float* mxB    = alloc((size_t)Nn_ * EHh_);
  float* denB   = alloc((size_t)Nn_ * EHh_);
  float* alphaB = alloc((size_t)(Ee_ + Nn_) * EHh_);
  float* degB   = alloc((size_t)Nn_);
  float* latB   = alloc((size_t)Nn_ * 4);
  float* escB   = alloc(64);
  // decoder view (reuses encoder region)
  o = obase;
  float* xdB = alloc((size_t)Bb_ * Ll_ * Dd_);
  float* xnB = alloc((size_t)Bb_ * Ll_ * Dd_);
  float* qB  = alloc((size_t)Bb_ * Ll_ * Dd_);
  float* kB  = alloc((size_t)Bb_ * Ll_ * Dd_);
  float* vB  = alloc((size_t)Bb_ * Ll_ * Dd_);
  float* oB  = alloc((size_t)Bb_ * Ll_ * Dd_);
  float* cvB = alloc((size_t)Bb_ * Dd_);
  float* coB = alloc((size_t)Bb_ * Dd_);
  float* ffmB    = qB;   // [B*L,DFF] overlays q+k (dead at FF time)
  float* logitsB = oB;   // overlays attn-out (dead after last layer)

  auto fill = [&](float* p, float v, long n) {
    k_fill<<<cdiv(n, 256), 256, 0, stream>>>(p, v, n);
  };
  // swizzle weight to bf16 fragment order once, then WMMA GEMM
  auto gemm = [&](const float* A, const float* Wm, const float* bias, float* Cm,
                  int M, int Nc, int K, int act, bool accum, bool nguard = false) {
    int NcPad = ((Nc + 63) / 64) * 64;
    long wn = (long)NcPad * K;
    k_wswz<<<cdiv(wn, 256), 256, 0, stream>>>(Wm, wbfB, K, Nc, NcPad);
    dim3 grid(NcPad / 64, M / 64);
    if (nguard)         k_gemm<0, false, true ><<<grid, 256, 0, stream>>>(A, wbfB, bias, Cm, M, Nc, K);
    else if (accum)     k_gemm<0, true , false><<<grid, 256, 0, stream>>>(A, wbfB, bias, Cm, M, Nc, K);
    else if (act == 1)  k_gemm<1, false, false><<<grid, 256, 0, stream>>>(A, wbfB, bias, Cm, M, Nc, K);
    else                k_gemm<0, false, false><<<grid, 256, 0, stream>>>(A, wbfB, bias, Cm, M, Nc, K);
  };

  // GAT message passing (H=16) into gout (pre-bias)
  auto run_gat16 = [&](const float* xin, const float* Wl, const float* asrc,
                       const float* adst, const float* Wel, const float* aedge,
                       float* gout) {
    gemm(xin, Wl, nullptr, xpB, Nn_, Dd_, Dd_, 0, false);
    k_heads_sd<<<Nn_, 512, 0, stream>>>(xpB, asrc, adst, sB, dB);
    k_escore<<<1, 64, 0, stream>>>(Wel, aedge, escB);
    fill(mxB, -1e30f, (long)Nn_ * EHh_);
    fill(denB, 0.f, (long)Nn_ * EHh_);
    fill(gout, 0.f, (long)Nn_ * Dd_);
    long tot = (long)(Ee_ + Nn_) * EHh_;
    k_alpha <<<cdiv(tot, 256), 256, 0, stream>>>(sB, dB, srcI, dstI, eattr, escB, latB, alphaB, EHh_, Ee_, tot);
    k_segmax<<<cdiv(tot, 256), 256, 0, stream>>>(alphaB, dstI, mxB, EHh_, Ee_, tot);
    k_expsum<<<cdiv(tot, 256), 256, 0, stream>>>(alphaB, dstI, mxB, denB, EHh_, Ee_, tot);
    k_scatter<<<Ee_ + Nn_, 256, 0, stream>>>(alphaB, denB, srcI, dstI, xpB, gout, EHh_, Ee_);
  };

  // ===================== encoder =====================
  // degree / loop edge-attr (layer invariant)
  fill(degB, 0.f, Nn_);
  fill(latB, 0.f, (long)Nn_ * 4);
  k_deg<<<cdiv(Ee_, 256), 256, 0, stream>>>(dstI, eattr, degB, latB, Ee_);
  k_loopattr<<<cdiv((long)Nn_ * 4, 256), 256, 0, stream>>>(latB, degB, (long)Nn_ * 4);

  // initial GAT on one-hot node features (H=1): xp = ne.W[feat]
  k_ne_gather<<<Nn_, 256, 0, stream>>>(nodef, neW, neas, nead, xpB, sB, dB);
  fill(mxB, -1e30f, Nn_); fill(denB, 0.f, Nn_); fill(gatoB, 0.f, (long)Nn_ * Dd_);
  {
    long tot = (long)(Ee_ + Nn_);
    k_alpha <<<cdiv(tot, 256), 256, 0, stream>>>(sB, dB, srcI, dstI, eattr, nullptr, latB, alphaB, 1, Ee_, tot);
    k_segmax<<<cdiv(tot, 256), 256, 0, stream>>>(alphaB, dstI, mxB, 1, Ee_, tot);
    k_expsum<<<cdiv(tot, 256), 256, 0, stream>>>(alphaB, dstI, mxB, denB, 1, Ee_, tot);
    k_scatter<<<Ee_ + Nn_, 256, 0, stream>>>(alphaB, denB, srcI, dstI, xpB, gatoB, 1, Ee_);
  }
  k_bias_act<<<cdiv((long)Nn_ * Dd_, 256), 256, 0, stream>>>(gatoB, neb, nfB, (long)Nn_ * Dd_);
  hipMemcpyAsync(denseB, nfB, (size_t)Nn_ * Dd_ * sizeof(float),
                 hipMemcpyDeviceToDevice, stream);

  const long NDn = (long)Nn_ * Dd_;
  for (int l = 0; l < NL_; ++l) {
    const float* lng = enc_lng + (long)l * 3 * Dd_;
    const float* lnb = enc_lnb + (long)l * 3 * Dd_;
    // h = nf + leaky(LN(gat1(nf) + b1))
    run_gat16(nfB, g1W + (long)l * Dd_ * Dd_, g1as + (long)l * EHh_ * 32,
              g1ad + (long)l * EHh_ * 32, g1We + (long)l * 4 * Dd_,
              g1ae + (long)l * EHh_ * 32, gatoB);
    k_ln<2, 1><<<Nn_, 256, 0, stream>>>(gatoB, g1b + (long)l * Dd_, lng, lnb, nfB, hB);
    // h = h + leaky(LN(gat2(h) + b2))
    run_gat16(hB, g2W + (long)l * Dd_ * Dd_, g2as + (long)l * EHh_ * 32,
              g2ad + (long)l * EHh_ * 32, g2We + (long)l * 4 * Dd_,
              g2ae + (long)l * EHh_ * 32, gatoB);
    k_ln<2, 1><<<Nn_, 256, 0, stream>>>(gatoB, g2b + (long)l * Dd_, lng + Dd_, lnb + Dd_, hB, hB);
    // h = h + leaky(LN(FFN(h)))
    gemm(hB, enc_fw1 + (long)l * Dd_ * DFF_, enc_fb1 + (long)l * DFF_, t0B, Nn_, DFF_, Dd_, 1, false);
    gemm(t0B, enc_fw2 + (long)l * DFF_ * Dd_, enc_fb2 + (long)l * Dd_, gatoB, Nn_, Dd_, DFF_, 0, false);
    k_ln<2, 1><<<Nn_, 256, 0, stream>>>(gatoB, nullptr, lng + 2 * Dd_, lnb + 2 * Dd_, hB, hB);
    k_carry<<<cdiv(NDn, 256), 256, 0, stream>>>(nfB, denseB, hB, NDn);
  }
  k_ln<0, 0><<<Nn_, 256, 0, stream>>>(nfB, nullptr, encLNg, encLNb, nullptr, nfB);
  k_pool<<<Bb_, 256, 0, stream>>>(nfB, poolB);
  gemm(poolB, muW, mub, muOut, Bb_, Dd_, Dd_, 0, false);
  gemm(poolB, sgW, sgb, sgOut, Bb_, Dd_, Dd_, 0, false);
  k_z<<<cdiv((long)Bb_ * Dd_, 256), 256, 0, stream>>>(muOut, sgOut, eps, zB, (long)Bb_ * Dd_);

  // ===================== decoder =====================
  const int BL = Bb_ * Ll_;
  const long BLD = (long)BL * Dd_;
  k_dec_init<<<cdiv(BLD, 256), 256, 0, stream>>>(smi, embW, xdB);

  for (int l = 0; l < NL_; ++l) {
    const float* lng = dec_lng + (long)l * 3 * Dd_;
    const float* lnb = dec_lnb + (long)l * 3 * Dd_;
    const float* saW = dec_saW + (long)l * 4 * Dd_ * Dd_;
    const float* sab = dec_sab + (long)l * 4 * Dd_;
    const float* caW = dec_caW + (long)l * 4 * Dd_ * Dd_;
    const float* cab = dec_cab + (long)l * 4 * Dd_;
    // causal self-attention
    k_ln<0, 0><<<BL, 256, 0, stream>>>(xdB, nullptr, lng, lnb, nullptr, xnB);
    gemm(xnB, saW + 0 * Dd_ * Dd_, sab + 0 * Dd_, qB, BL, Dd_, Dd_, 0, false);
    gemm(xnB, saW + 1 * Dd_ * Dd_, sab + 1 * Dd_, kB, BL, Dd_, Dd_, 0, false);
    gemm(xnB, saW + 2 * Dd_ * Dd_, sab + 2 * Dd_, vB, BL, Dd_, Dd_, 0, false);
    k_attn<<<(Bb_ * DHh_ * Ll_) / 8, 256, 0, stream>>>(qB, kB, vB, oB);
    gemm(oB, saW + 3 * Dd_ * Dd_, sab + 3 * Dd_, xdB, BL, Dd_, Dd_, 0, true);
    // cross-attention: 1 memory token => softmax == 1 => O = (z@W2+b2)@W3+b3 broadcast
    gemm(zB,  caW + 2 * Dd_ * Dd_, cab + 2 * Dd_, cvB, Bb_, Dd_, Dd_, 0, false);
    gemm(cvB, caW + 3 * Dd_ * Dd_, cab + 3 * Dd_, coB, Bb_, Dd_, Dd_, 0, false);
    k_bcastadd<<<cdiv(BLD, 256), 256, 0, stream>>>(xdB, coB, BLD);
    // FFN
    k_ln<0, 0><<<BL, 256, 0, stream>>>(xdB, nullptr, lng + 2 * Dd_, lnb + 2 * Dd_, nullptr, xnB);
    gemm(xnB, dec_fw1 + (long)l * Dd_ * DFF_, dec_fb1 + (long)l * DFF_, ffmB, BL, DFF_, Dd_, 1, false);
    gemm(ffmB, dec_fw2 + (long)l * DFF_ * Dd_, dec_fb2 + (long)l * Dd_, xdB, BL, Dd_, DFF_, 0, true);
  }
  k_ln<0, 0><<<BL, 256, 0, stream>>>(xdB, nullptr, decLNg, decLNb, nullptr, xnB);
  gemm(xnB, genW, genb, logitsB, BL, SMIV_, Dd_, 0, false);
  k_lsm<<<BL, 128, 0, stream>>>(logitsB, outp);
}
